// MLPNetwork_3221225472053
// MI455X (gfx1250) — matrix operand order repaired and checked by reading, not compiled, not run on hardware
//
#include <hip/hip_runtime.h>

typedef __bf16 bf16_t;
typedef __attribute__((ext_vector_type(16))) bf16_t v16bf;
typedef __attribute__((ext_vector_type(8)))  float  v8f;

#define IN_DIM   640
#define HD       64
#define M_TOTAL  262169
#define NSHOT    25
#define NBLK     2049          // ceil(M_TOTAL / 128)
#define KCH1     20            // 640 / 32
#define KCH2     2             // 64 / 32
#define BN_EPS   1e-5f

// workspace layout (float/dword units)
#define WS_STATS_A 0
#define WS_STATS_B 128
#define WS_BPACK1  256
#define WS_BPACK2  (WS_BPACK1 + KCH1*4*256)      // 20736
#define WS_PART_A  (WS_BPACK2 + KCH2*4*256)      // 22784
#define WS_PART_B  (WS_PART_A + NBLK*128)        // 285056
#define WS_Y1H     (WS_PART_B + NBLK*128)        // 547328 ; bf16 Y1 lives here

struct B32x8 { unsigned u[8]; };

// precise RNE pack (cold paths: weight pack)
__device__ __forceinline__ unsigned f2bf_rne(float f) {
  unsigned u = __float_as_uint(f);
  return ((u + 0x7FFFu + ((u >> 16) & 1u)) >> 16) & 0xFFFFu;
}
__device__ __forceinline__ unsigned pk2_rne(float a, float b) {
  return f2bf_rne(a) | (f2bf_rne(b) << 16);
}
// fast pack (hot paths): round-half-away via +0x8000, merge high halves
__device__ __forceinline__ unsigned pk_hi(unsigned a, unsigned b) {
  return (a >> 16) | (b & 0xFFFF0000u);   // candidates for v_perm_b32
}
__device__ __forceinline__ unsigned pkrn_u(unsigned a, unsigned b) {
  return pk_hi(a + 0x8000u, b + 0x8000u);
}
__device__ __forceinline__ unsigned pkrn_f(float a, float b) {
  return pkrn_u(__float_as_uint(a), __float_as_uint(b));
}
__device__ __forceinline__ unsigned short bf_store(float v) {
  return (unsigned short)((__float_as_uint(v) + 0x8000u) >> 16);
}
__device__ __forceinline__ float bf2f(unsigned h) { return __uint_as_float(h << 16); }

__device__ __forceinline__ v16bf load_b16x16(const unsigned* p) {
  const uint4* q = reinterpret_cast<const uint4*>(p);
  uint4 q0 = q[0], q1 = q[1];
  B32x8 s = {q0.x, q0.y, q0.z, q0.w, q1.x, q1.y, q1.z, q1.w};
  return __builtin_bit_cast(v16bf, s);
}

// ---------------------------------------------------------------- weight pack
// B-matrix bf16 layout (wave32, 32x16 K x N): lane -> N = lane&15;
// lanes 0-15 hold K 0-7,16-23; lanes 16-31 hold K 8-15,24-31 (VGPR v: 2 consecutive K).
__global__ void pack_w_kernel(const float* __restrict__ W1,
                              const float* __restrict__ W2,
                              float* __restrict__ ws) {
  int tid = blockIdx.x * blockDim.x + threadIdx.x;
  const int n1 = KCH1 * 4 * 256;
  const int total = n1 + KCH2 * 4 * 256;
  if (tid >= total) return;
  const float* W; unsigned* dst; int idx;
  if (tid < n1) { W = W1; dst = (unsigned*)(ws + WS_BPACK1); idx = tid; }
  else          { W = W2; dst = (unsigned*)(ws + WS_BPACK2); idx = tid - n1; }
  int c = idx / 1024, rem = idx % 1024;
  int t = rem / 256,  rem2 = rem % 256;
  int l = rem2 / 8,   d = rem2 % 8;
  int k = c * 32 + ((l < 16) ? 0 : 8) + ((d < 4) ? 2 * d : 16 + 2 * (d - 4));
  int n = t * 16 + (l & 15);
  dst[idx] = pk2_rne(W[(size_t)k * HD + n], W[(size_t)(k + 1) * HD + n]);
}

// ---------------------------------------------------------------- GEMM1 (bf16 WMMA)
__global__ __launch_bounds__(256) void gemm1_kernel(const float* __restrict__ xs,
                                                    const float* __restrict__ xq,
                                                    float* __restrict__ ws) {
  __shared__ float lsum[64], lsq[64];
  int tid = threadIdx.x;
  if (tid < 64) { lsum[tid] = 0.f; lsq[tid] = 0.f; }
  __syncthreads();
  int wave = tid >> 5, lane = tid & 31;
  int rl = lane & 15, half = lane >> 4;
  int row0 = blockIdx.x * 128 + wave * 16;
  int r = row0 + rl; if (r > M_TOTAL - 1) r = M_TOTAL - 1;  // clamp: EXEC stays all-1s
  const float* rowp = (r < NSHOT) ? (xs + (size_t)r * IN_DIM)
                                  : (xq + (size_t)(r - NSHOT) * IN_DIM);
  const unsigned* rowu = (const unsigned*)rowp;
  const unsigned* bp = (const unsigned*)(ws + WS_BPACK1);
  v8f acc0 = {}, acc1 = {}, acc2 = {}, acc3 = {};
  // software pipeline: preload chunk 0's A
  int base = half * 8;
  uint4 q0 = *(const uint4*)(rowu + base);
  uint4 q1 = *(const uint4*)(rowu + base + 4);
  uint4 q2 = *(const uint4*)(rowu + base + 16);
  uint4 q3 = *(const uint4*)(rowu + base + 20);
  for (int c = 0; c < KCH1; ++c) {
    // A layout: VGPR0-3 <- K base..base+7, VGPR4-7 <- K base+16..base+23
    B32x8 as;
    as.u[0] = pkrn_u(q0.x, q0.y); as.u[1] = pkrn_u(q0.z, q0.w);
    as.u[2] = pkrn_u(q1.x, q1.y); as.u[3] = pkrn_u(q1.z, q1.w);
    as.u[4] = pkrn_u(q2.x, q2.y); as.u[5] = pkrn_u(q2.z, q2.w);
    as.u[6] = pkrn_u(q3.x, q3.y); as.u[7] = pkrn_u(q3.z, q3.w);
    v16bf a = __builtin_bit_cast(v16bf, as);
    // B loads first (in-order VMEM return: B must come back before next-A)
    const unsigned* bc = bp + (size_t)(c * 4) * 256 + lane * 8;
    v16bf b0 = load_b16x16(bc);
    v16bf b1 = load_b16x16(bc + 256);
    v16bf b2 = load_b16x16(bc + 512);
    v16bf b3 = load_b16x16(bc + 768);
    // then issue next chunk's A loads; they stay in flight under the WMMAs
    if (c + 1 < KCH1) {
      int nb = (c + 1) * 32 + half * 8;
      q0 = *(const uint4*)(rowu + nb);
      q1 = *(const uint4*)(rowu + nb + 4);
      q2 = *(const uint4*)(rowu + nb + 16);
      q3 = *(const uint4*)(rowu + nb + 20);
    }
    acc0 = __builtin_amdgcn_wmma_f32_16x16x32_bf16(false, a, false, b0, (short)0, acc0, false, false);
    acc1 = __builtin_amdgcn_wmma_f32_16x16x32_bf16(false, a, false, b1, (short)0, acc1, false, false);
    acc2 = __builtin_amdgcn_wmma_f32_16x16x32_bf16(false, a, false, b2, (short)0, acc2, false, false);
    acc3 = __builtin_amdgcn_wmma_f32_16x16x32_bf16(false, a, false, b3, (short)0, acc3, false, false);
  }
  // C layout: (VGPR i, lane) -> row = row0 + i + 8*(lane>>4), col = t*16 + (lane&15)
  unsigned short* y1h = (unsigned short*)(ws + WS_Y1H);
  v8f accs[4] = {acc0, acc1, acc2, acc3};
#pragma unroll
  for (int t = 0; t < 4; ++t) {
    int colg = t * 16 + rl;
    float s = 0.f, s2 = 0.f;
#pragma unroll
    for (int i = 0; i < 8; ++i) {
      int rowe = row0 + i + 8 * half;
      float v = accs[t][i];
      if (rowe < M_TOTAL) {
        y1h[(size_t)rowe * HD + colg] = bf_store(v);
        s += v; s2 += v * v;
      }
    }
    atomicAdd(&lsum[colg], s);
    atomicAdd(&lsq[colg], s2);
  }
  __syncthreads();
  if (tid < 64) {
    ws[WS_PART_A + (size_t)blockIdx.x * 128 + tid] = lsum[tid];
    ws[WS_PART_A + (size_t)blockIdx.x * 128 + 64 + tid] = lsq[tid];
  }
}

// ---------------------------------------------------------------- stats reduce
__global__ void reduce_stats_kernel(float* __restrict__ ws, int srcOff, int dstOff) {
  int tid = threadIdx.x;  // 128 threads
  float s = 0.f;
  for (int b = 0; b < NBLK; ++b) s += ws[srcOff + (size_t)b * 128 + tid];
  ws[dstOff + tid] = s;
}

// ---------------------------------------------------------------- GEMM2 (BN1+ReLU fused)
__global__ __launch_bounds__(256) void gemm2_kernel(const float* __restrict__ g1,
                                                    const float* __restrict__ be1,
                                                    float* __restrict__ ws,
                                                    float* __restrict__ out) {
  __shared__ float sc[64], sh[64], lsum[64], lsq[64];
  int tid = threadIdx.x;
  if (tid < 64) {
    float mean = ws[WS_STATS_A + tid] * (1.0f / M_TOTAL);
    float var  = ws[WS_STATS_A + 64 + tid] * (1.0f / M_TOTAL) - mean * mean;
    float s = g1[tid] * rsqrtf(var + BN_EPS);
    sc[tid] = s; sh[tid] = be1[tid] - mean * s;
    lsum[tid] = 0.f; lsq[tid] = 0.f;
  }
  __syncthreads();
  int wave = tid >> 5, lane = tid & 31;
  int rl = lane & 15, half = lane >> 4;
  int row0 = blockIdx.x * 128 + wave * 16;
  int r = row0 + rl; if (r > M_TOTAL - 1) r = M_TOTAL - 1;
  const unsigned short* y1h = (const unsigned short*)(ws + WS_Y1H);
  const unsigned* bp = (const unsigned*)(ws + WS_BPACK2);
  v8f acc0 = {}, acc1 = {}, acc2 = {}, acc3 = {};
#pragma unroll
  for (int c = 0; c < KCH2; ++c) {
    int base = c * 32 + half * 8;
    uint4 q0 = *(const uint4*)(y1h + (size_t)r * HD + base);
    uint4 q1 = *(const uint4*)(y1h + (size_t)r * HD + base + 16);
    unsigned rv[8] = {q0.x, q0.y, q0.z, q0.w, q1.x, q1.y, q1.z, q1.w};
    B32x8 as;
#pragma unroll
    for (int j = 0; j < 8; ++j) {
      int k = base + ((j < 4) ? 2 * j : 16 + 2 * (j - 4));
      float v0 = fmaxf(0.f, bf2f(rv[j] & 0xFFFFu) * sc[k] + sh[k]);
      float v1 = fmaxf(0.f, bf2f(rv[j] >> 16) * sc[k + 1] + sh[k + 1]);
      as.u[j] = pkrn_f(v0, v1);
    }
    v16bf a = __builtin_bit_cast(v16bf, as);
    const unsigned* bc = bp + (size_t)(c * 4) * 256 + lane * 8;
    v16bf b0 = load_b16x16(bc);
    v16bf b1 = load_b16x16(bc + 256);
    v16bf b2 = load_b16x16(bc + 512);
    v16bf b3 = load_b16x16(bc + 768);
    acc0 = __builtin_amdgcn_wmma_f32_16x16x32_bf16(false, a, false, b0, (short)0, acc0, false, false);
    acc1 = __builtin_amdgcn_wmma_f32_16x16x32_bf16(false, a, false, b1, (short)0, acc1, false, false);
    acc2 = __builtin_amdgcn_wmma_f32_16x16x32_bf16(false, a, false, b2, (short)0, acc2, false, false);
    acc3 = __builtin_amdgcn_wmma_f32_16x16x32_bf16(false, a, false, b3, (short)0, acc3, false, false);
  }
  float* o = out + 320;  // feat rows (shot then query) are contiguous here
  v8f accs[4] = {acc0, acc1, acc2, acc3};
#pragma unroll
  for (int t = 0; t < 4; ++t) {
    int colg = t * 16 + rl;
    float s = 0.f, s2 = 0.f;
#pragma unroll
    for (int i = 0; i < 8; ++i) {
      int rowe = row0 + i + 8 * half;
      float v = accs[t][i];
      if (rowe < M_TOTAL) {
        o[(size_t)rowe * HD + colg] = v;  // raw Y2; BN2 applied in next pass
        s += v; s2 += v * v;
      }
    }
    atomicAdd(&lsum[colg], s);
    atomicAdd(&lsq[colg], s2);
  }
  __syncthreads();
  if (tid < 64) {
    ws[WS_PART_B + (size_t)blockIdx.x * 128 + tid] = lsum[tid];
    ws[WS_PART_B + (size_t)blockIdx.x * 128 + 64 + tid] = lsq[tid];
  }
}

// ---------------------------------------------------------------- BN2+ReLU in-place
__global__ __launch_bounds__(256) void bn2_kernel(const float* __restrict__ g2,
                                                  const float* __restrict__ be2,
                                                  const float* __restrict__ ws,
                                                  float* __restrict__ out) {
  __shared__ float sc[64], sh[64];
  int tid = threadIdx.x;
  if (tid < 64) {
    float mean = ws[WS_STATS_B + tid] * (1.0f / M_TOTAL);
    float var  = ws[WS_STATS_B + 64 + tid] * (1.0f / M_TOTAL) - mean * mean;
    float s = g2[tid] * rsqrtf(var + BN_EPS);
    sc[tid] = s; sh[tid] = be2[tid] - mean * s;
  }
  __syncthreads();
  float4* p = (float4*)(out + 320);
  const size_t NV4 = (size_t)M_TOTAL * HD / 4;
  for (size_t i = (size_t)blockIdx.x * blockDim.x + tid; i < NV4;
       i += (size_t)gridDim.x * blockDim.x) {
    float4 v = p[i];
    int cb = (int)((i * 4) & 63);
    v.x = fmaxf(0.f, v.x * sc[cb]     + sh[cb]);
    v.y = fmaxf(0.f, v.y * sc[cb + 1] + sh[cb + 1]);
    v.z = fmaxf(0.f, v.z * sc[cb + 2] + sh[cb + 2]);
    v.w = fmaxf(0.f, v.w * sc[cb + 3] + sh[cb + 3]);
    p[i] = v;
  }
}

// ---------------------------------------------------------------- JAX threefry noise
__device__ __forceinline__ float erfinv_f(float x) {
  float w = -__logf(fmaf(x, -x, 1.0f)), p;
  if (w < 5.0f) {
    w -= 2.5f;
    p = 2.81022636e-08f;
    p = fmaf(p, w, 3.43273939e-07f);  p = fmaf(p, w, -3.5233877e-06f);
    p = fmaf(p, w, -4.39150654e-06f); p = fmaf(p, w, 0.00021858087f);
    p = fmaf(p, w, -0.00125372503f);  p = fmaf(p, w, -0.00417768164f);
    p = fmaf(p, w, 0.246640727f);     p = fmaf(p, w, 1.50140941f);
  } else {
    w = __fsqrt_rn(w) - 3.0f;
    p = -0.000200214257f;
    p = fmaf(p, w, 0.000100950558f);  p = fmaf(p, w, 0.00134934322f);
    p = fmaf(p, w, -0.00367342844f);  p = fmaf(p, w, 0.00573950773f);
    p = fmaf(p, w, -0.0076224613f);   p = fmaf(p, w, 0.00943887047f);
    p = fmaf(p, w, 1.00167406f);      p = fmaf(p, w, 2.83297682f);
  }
  return p * x;
}

#define TF_R(a) { x0 += x1; x1 = (x1 << (a)) | (x1 >> (32 - (a))); x1 ^= x0; }
__device__ float jax_normal(unsigned j) {  // element j of normal(key(123),(9600,))
  const unsigned HALF = 4800u;
  unsigned idx = (j < HALF) ? j : j - HALF;
  unsigned x0 = idx, x1 = idx + HALF;
  const unsigned k0 = 0u, k1 = 123u, k2 = k0 ^ k1 ^ 0x1BD11BDAu;
  x0 += k0; x1 += k1;
  TF_R(13) TF_R(15) TF_R(26) TF_R(6)  x0 += k1; x1 += k2 + 1u;
  TF_R(17) TF_R(29) TF_R(16) TF_R(24) x0 += k2; x1 += k0 + 2u;
  TF_R(13) TF_R(15) TF_R(26) TF_R(6)  x0 += k0; x1 += k1 + 3u;
  TF_R(17) TF_R(29) TF_R(16) TF_R(24) x0 += k1; x1 += k2 + 4u;
  TF_R(13) TF_R(15) TF_R(26) TF_R(6)  x0 += k2; x1 += k0 + 5u;
  unsigned bits = (j < HALF) ? x0 : x1;
  float u01 = __uint_as_float((bits >> 9) | 0x3f800000u) - 1.0f;
  const float lo = -0.99999994f;
  float u = fmaxf(lo, fmaf(u01, 1.0f - lo, lo));
  return 1.41421356f * erfinv_f(u);
}

// ---------------------------------------------------------------- head: proto->mu/logvar->z
__global__ void head_kernel(const float* __restrict__ W3a, const float* __restrict__ g3a,
                            const float* __restrict__ be3a, const float* __restrict__ W3b,
                            const float* __restrict__ g3b, const float* __restrict__ be3b,
                            float* __restrict__ out) {
  __shared__ float proto[5][64];
  int n = threadIdx.x;  // 64 threads, one column each
  const float* fs = out + 320;
  for (int p = 0; p < 5; ++p) {
    float s = 0.f;
    for (int j = 0; j < 5; ++j) s += fs[(size_t)(5 * p + j) * HD + n];
    proto[p][n] = s * 0.2f;
  }
  __syncthreads();
  float pa[5], pb[5];
  for (int p = 0; p < 5; ++p) {
    float sa = 0.f, sb = 0.f;
    for (int k = 0; k < 64; ++k) {
      float pv = proto[p][k];
      sa = fmaf(pv, W3a[(size_t)k * HD + n], sa);
      sb = fmaf(pv, W3b[(size_t)k * HD + n], sb);
    }
    pa[p] = sa; pb[p] = sb;   // biases b3a/b3b cancel in BN
  }
  float ma = 0.f, mb = 0.f;
  for (int p = 0; p < 5; ++p) { ma += pa[p]; mb += pb[p]; }
  ma *= 0.2f; mb *= 0.2f;
  float va = 0.f, vb = 0.f;
  for (int p = 0; p < 5; ++p) {
    float da = pa[p] - ma, db = pb[p] - mb;
    va += da * da; vb += db * db;
  }
  float ra = rsqrtf(va * 0.2f + BN_EPS), rb = rsqrtf(vb * 0.2f + BN_EPS);
  for (int p = 0; p < 5; ++p) {
    float mu = fmaxf(0.f, g3a[n] * (pa[p] - ma) * ra + be3a[n]);
    float lv = fmaxf(0.f, g3b[n] * (pb[p] - mb) * rb + be3b[n]);
    float sd = 0.5f * __expf(lv);
    float es = 0.f;
    for (int smp = 0; smp < 30; ++smp)
      es += jax_normal((unsigned)(smp * 320 + p * 64 + n));
    out[p * 64 + n] = mu + sd * (es * (1.0f / 30.0f));
  }
}

// ---------------------------------------------------------------- launch
extern "C" void kernel_launch(void* const* d_in, const int* in_sizes, int n_in,
                              void* d_out, int out_size, void* d_ws, size_t ws_size,
                              hipStream_t stream) {
  (void)in_sizes; (void)n_in; (void)out_size; (void)ws_size;
  const float* xs  = (const float*)d_in[0];
  const float* xq  = (const float*)d_in[1];
  const float* W1  = (const float*)d_in[2];
  const float* g1  = (const float*)d_in[4];
  const float* be1 = (const float*)d_in[5];
  const float* W2  = (const float*)d_in[6];
  const float* g2  = (const float*)d_in[8];
  const float* be2 = (const float*)d_in[9];
  const float* W3a = (const float*)d_in[10];
  const float* g3a = (const float*)d_in[12];
  const float* be3a= (const float*)d_in[13];
  const float* W3b = (const float*)d_in[14];
  const float* g3b = (const float*)d_in[16];
  const float* be3b= (const float*)d_in[17];
  float* ws  = (float*)d_ws;
  float* out = (float*)d_out;

  pack_w_kernel<<<88, 256, 0, stream>>>(W1, W2, ws);
  gemm1_kernel<<<NBLK, 256, 0, stream>>>(xs, xq, ws);
  reduce_stats_kernel<<<1, 128, 0, stream>>>(ws, WS_PART_A, WS_STATS_A);
  gemm2_kernel<<<NBLK, 256, 0, stream>>>(g1, be1, ws, out);
  reduce_stats_kernel<<<1, 128, 0, stream>>>(ws, WS_PART_B, WS_STATS_B);
  bn2_kernel<<<4096, 256, 0, stream>>>(g2, be2, ws, out);
  head_kernel<<<1, 64, 0, stream>>>(W3a, g3a, be3a, W3b, g3b, be3b, out);
}
